// AssembledBTBAcMasterModel_47854525612408
// MI455X (gfx1250) — compile-verified, matrix operand order checked by reading, not run on hardware
//
#include <hip/hip_runtime.h>
#include <hip/hip_bf16.h>
#include <stdint.h>

// ---------------------------------------------------------------------------
// Graph random-walk (B=8192 start nodes, P=32 paths, S=16 steps, D=64 degree)
// MI455X / gfx1250: wave32; WMMA bf16 column-sum reduction (products in B,
// ones in A => every D vgpr holds the per-walker sum, no extraction cost);
// constexpr threefry step keys; wave-uniform all-dead early exit; rcp-based
// sigmoid; tables (77MB) are L2-resident on the 192MB L2 -> gather-bound.
// ---------------------------------------------------------------------------

#define MAX_STEPS 16
#define NUM_PATHS 32
#define DEG       64

typedef __bf16 bf16_t;
typedef __attribute__((ext_vector_type(16))) __bf16 v16bf;
typedef __attribute__((ext_vector_type(8)))  float  v8f;

struct KP { uint32_t a, b; };

// threefry-2x32/20, usable at compile time (step keys) and runtime (per-path).
__host__ __device__ constexpr KP tf2x32(uint32_t k0, uint32_t k1,
                                        uint32_t x0, uint32_t x1) {
    uint32_t ks[3] = {k0, k1, k0 ^ k1 ^ 0x1BD11BDAu};
    x0 += ks[0];
    x1 += ks[1];
    const uint32_t R[2][4] = {{13u, 15u, 26u, 6u}, {17u, 29u, 16u, 24u}};
    for (int g = 0; g < 5; ++g) {
        for (int j = 0; j < 4; ++j) {
            x0 += x1;
            const uint32_t r = R[g & 1][j];
            x1 = (x1 << r) | (x1 >> (32u - r));
            x1 ^= x0;
        }
        x0 += ks[(g + 1) % 3];
        x1 += ks[(g + 2) % 3] + (uint32_t)(g + 1);
    }
    return {x0, x1};
}

// key = jax.random.fold_in(jax.random.key(42), step)  -- compile-time constants
constexpr KP STEP_KEYS[MAX_STEPS] = {
    tf2x32(0u, 42u, 0u, 0u),  tf2x32(0u, 42u, 0u, 1u),
    tf2x32(0u, 42u, 0u, 2u),  tf2x32(0u, 42u, 0u, 3u),
    tf2x32(0u, 42u, 0u, 4u),  tf2x32(0u, 42u, 0u, 5u),
    tf2x32(0u, 42u, 0u, 6u),  tf2x32(0u, 42u, 0u, 7u),
    tf2x32(0u, 42u, 0u, 8u),  tf2x32(0u, 42u, 0u, 9u),
    tf2x32(0u, 42u, 0u, 10u), tf2x32(0u, 42u, 0u, 11u),
    tf2x32(0u, 42u, 0u, 12u), tf2x32(0u, 42u, 0u, 13u),
    tf2x32(0u, 42u, 0u, 14u), tf2x32(0u, 42u, 0u, 15u)};

__global__ __launch_bounds__(256)
void gw_walk_kernel(const int*   __restrict__ adj,
                    const float* __restrict__ tau,
                    const float* __restrict__ alpha,
                    const float* __restrict__ quality,
                    const int*   __restrict__ start,
                    const float* __restrict__ phi1,
                    const float* __restrict__ phi2,
                    float*       __restrict__ out,
                    int B) {
    const int lane = (int)(threadIdx.x & 31u);
    const int wave = (int)((blockIdx.x * blockDim.x + threadIdx.x) >> 5);
    const int m    = lane & 15;          // walker within wave (0..15)
    const int half = (lane >> 4) & 1;    // which contiguous K-half this lane owns
    const int path = wave * 16 + m;      // global walker id in [0, B*P)
    const int BP   = B * NUM_PATHS;

    const float p1 = phi1[0];
    const float p2 = phi2[0];

    int   cur    = start[path >> 5];     // path / NUM_PATHS
    float energy = 1.0f;

    // step 0 outputs (init node / energy)
    if (half == 0) {
        out[path] = (float)cur;
        out[(size_t)MAX_STEPS * (size_t)BP + (size_t)path] = 1.0f;
    }

    // all-ones A matrix (16x32 bf16): D[M][N] = colsum_N(B) for every M,
    // so every D vgpr in every lane holds walker (lane%16)'s row sum.
    v16bf aones;
#pragma unroll
    for (int e = 0; e < 16; ++e) aones[e] = (bf16_t)1.0f;

    for (int step = 1; step < MAX_STEPS; ++step) {
        // ---- wave-uniform fast path: all 16 walkers dead -> outputs frozen ----
        if (__ballot(cur >= 0) == 0ull) {
            if (half == 0) {
                for (int s = step; s < MAX_STEPS; ++s) {
                    out[(size_t)s * (size_t)BP + (size_t)path] = -1.0f;
                    out[(size_t)(MAX_STEPS + s) * (size_t)BP + (size_t)path] = 0.0f;
                }
            }
            break;
        }

        const bool valid = (cur >= 0);
        const int  safe  = valid ? cur : 0;
        const int*   arow = adj   + (size_t)safe * DEG;
        const float* trow = tau   + (size_t)safe * DEG;
        const float* frow = alpha + (size_t)safe * DEG;

        // ---- B-matrix layout: lane owns 16 contiguous K per WMMA ----
        // B1: K = half*16 + e ; B2: K = 32 + half*16 + e   (e = 0..15)
        // NOTE: tau/alpha are pre-zeroed at padded positions in the input data,
        // so prod needs no validity mask; the compares only feed row_len.
        float prod[32];
        int   cnt = 0;
#pragma unroll
        for (int c2 = 0; c2 < 2; ++c2) {
            const int kbase = c2 * 32 + half * 16;   // 0/16 then 32/48
#pragma unroll
            for (int q4 = 0; q4 < 4; ++q4) {
                const int4   ia = *(const int4*)  (arow + kbase + q4 * 4);
                const float4 ta = *(const float4*)(trow + kbase + q4 * 4);
                const float4 fa = *(const float4*)(frow + kbase + q4 * 4);
                const int   av[4] = {ia.x, ia.y, ia.z, ia.w};
                const float tv[4] = {ta.x, ta.y, ta.z, ta.w};
                const float fv[4] = {fa.x, fa.y, fa.z, fa.w};
#pragma unroll
                for (int j = 0; j < 4; ++j) {
                    cnt += (av[j] >= 0) ? 1 : 0;
                    prod[c2 * 16 + q4 * 4 + j] = tv[j] * fv[j];
                }
            }
        }
        // valid out-degree of the 64-wide row (this lane's 32 + partner's 32)
        const int row_len = cnt + __shfl_xor(cnt, 16, 32);

        // ---- 64-wide sum(alpha*tau) for 16 walkers: 2 WMMAs, K=32 each ----
        v16bf b1, b2;
#pragma unroll
        for (int e = 0; e < 16; ++e) {
            b1[e] = (bf16_t)prod[e];
            b2[e] = (bf16_t)prod[16 + e];
        }
        v8f c = {0.f, 0.f, 0.f, 0.f, 0.f, 0.f, 0.f, 0.f};
        c = __builtin_amdgcn_wmma_f32_16x16x32_bf16(false, aones, false, b1,
                                                    (short)0, c, false, false);
        c = __builtin_amdgcn_wmma_f32_16x16x32_bf16(false, aones, false, b2,
                                                    (short)0, c, false, false);
        const float sum_at = c[0];   // colsum for walker lane%16, any M

        // ---- threefry RNG: precomputed step key + per-path counter block ----
        const KP kp = STEP_KEYS[step];
        const uint32_t i  = (uint32_t)path;
        const uint32_t hn = (uint32_t)(BP >> 1);
        const uint32_t c0 = (i < hn) ? i : (i - hn);
        const uint32_t c1 = (i < hn) ? (i + hn) : i;
        const KP ob = tf2x32(kp.a, kp.b, c0, c1);
        const uint32_t bits = (i < hn) ? ob.a : ob.b;
        const float u = __uint_as_float((bits >> 9) | 0x3f800000u) - 1.0f;

        int samp = (int)(u * (float)row_len);     // floor: operands >= 0
        samp = (samp < 0) ? 0 : ((samp > DEG - 1) ? (DEG - 1) : samp);
        const bool has  = valid && (row_len > 0);
        const int  sidx = has ? samp : 0;

        // point-gather sampled edge (L2-resident, 12B)
        const int   g_n = arow[sidx];
        const float g_t = trow[sidx];
        const float g_a = frow[sidx];

        int   nxt = has ? g_n : cur;
        const float nt = has ? g_t : 0.0f;
        const float na = has ? g_a : 0.0f;

        const float norm = (na * nt) * __builtin_amdgcn_rcpf(fmaxf(sum_at, 1e-9f));
        const float q    = quality[(nxt >= 0) ? nxt : 0];
        const float z    = p1 * norm + p2 * q;
        // sigmoid(-z) = 1/(1+e^z): one v_exp + one v_rcp, no IEEE divide chain
        const float drop = __builtin_amdgcn_rcpf(1.0f + __expf(z));

        energy -= drop;
        const bool alive = (energy > 0.0f);
        nxt    = alive ? nxt : -1;
        energy = alive ? energy : 0.0f;

        // prefetch next step's rows (256B each) while stores retire
        {
            const int pn = (nxt >= 0) ? nxt : 0;
            const size_t rb = (size_t)pn * DEG;
            __builtin_prefetch(adj   + rb, 0, 1);
            __builtin_prefetch(adj   + rb + 32, 0, 1);
            __builtin_prefetch(tau   + rb, 0, 1);
            __builtin_prefetch(tau   + rb + 32, 0, 1);
            __builtin_prefetch(alpha + rb, 0, 1);
            __builtin_prefetch(alpha + rb + 32, 0, 1);
        }

        if (half == 0) {
            out[(size_t)step * (size_t)BP + (size_t)path] = (float)nxt;
            out[(size_t)(MAX_STEPS + step) * (size_t)BP + (size_t)path] = energy;
        }
        cur = nxt;
    }
}

extern "C" void kernel_launch(void* const* d_in, const int* in_sizes, int n_in,
                              void* d_out, int out_size, void* d_ws, size_t ws_size,
                              hipStream_t stream) {
    (void)n_in; (void)out_size; (void)d_ws; (void)ws_size;
    const int*   adj     = (const int*)  d_in[0];
    const float* tau     = (const float*)d_in[1];
    const float* alpha   = (const float*)d_in[2];
    const float* quality = (const float*)d_in[3];
    const int*   start   = (const int*)  d_in[4];
    const float* phi1    = (const float*)d_in[5];
    const float* phi2    = (const float*)d_in[6];
    float* out = (float*)d_out;

    const int B  = in_sizes[4];          // 8192 start nodes
    const int BP = B * NUM_PATHS;        // 262144 walkers
    // one wave32 per 16 walkers -> 2 threads per walker
    const int threads_total = BP * 2;
    const int block = 256;
    const int grid  = (threads_total + block - 1) / block;

    gw_walk_kernel<<<grid, block, 0, stream>>>(adj, tau, alpha, quality, start,
                                               phi1, phi2, out, B);
}
